// ImputePerformer_2190433321797
// MI455X (gfx1250) — matrix operand order repaired and checked by reading, not compiled, not run on hardware
//
#include <hip/hip_runtime.h>
#include <hip/hip_bf16.h>

typedef __attribute__((ext_vector_type(16))) _Float16 v16h;
typedef __attribute__((ext_vector_type(8)))  _Float16 v8h;
typedef __attribute__((ext_vector_type(8)))  float    v8f;
typedef __attribute__((ext_vector_type(4)))  unsigned int u32x4;
typedef __attribute__((ext_vector_type(8)))  unsigned int u32x8;

namespace {
constexpr int kB = 8, kN = 4096, kDim = 64, kH = 8, kDH = 64, kInner = 512, kL = 2;
constexpr int kM = 266, kMP = 288;              // pad 266 -> 288 = 9*32
constexpr float kEps   = 1e-4f;
constexpr float kDn    = 0.35355339059327373f;  // 64^-0.25
constexpr float kRatio = 0.06131393394849658f;  // 266^-0.5
}

// ---------------- WMMA fragment helpers (ISA 7.12.2, wave32) ----------------

__device__ __forceinline__ v16h cat16(v8h lo, v8h hi) {
  return __builtin_shufflevector(lo, hi, 0, 1, 2, 3, 4, 5, 6, 7, 8, 9, 10, 11,
                                 12, 13, 14, 15);
}

// A[m][k] fragment from row-major f16 (two b128 loads; lda multiple of 8).
// Lane layout: row m=lane%16; g=lane/16: K = {16g..16g+15}.
__device__ __forceinline__ v16h load_a_rm(const _Float16* p, int lda, int lane) {
  int m = lane & 15, g = lane >> 4;
  const _Float16* r = p + (size_t)m * lda + 16 * g;
  return cat16(*(const v8h*)r, *(const v8h*)(r + 8));
}

// Fragment from pre-swizzled tile (32 lanes x 16 contiguous halves = 1KB).
__device__ __forceinline__ v16h load_pk(const _Float16* tile, int lane) {
  const _Float16* r = tile + lane * 16;
  return cat16(*(const v8h*)r, *(const v8h*)(r + 8));
}

__device__ __forceinline__ v8f wmma32(v16h a, v16h b, v8f c) {
  return __builtin_amdgcn_wmma_f32_16x16x32_f16(false, a, false, b, (short)0, c,
                                                false, false);
}

// ---------------- setup / pack kernels --------------------------------------

__global__ void k_init_h(const float* __restrict__ x, float* h32, _Float16* h16) {
  int idx = blockIdx.x * blockDim.x + threadIdx.x;
  if (idx >= kB * kN * kDim) return;
  float v = x[idx >> 6];
  h32[idx] = v;
  h16[idx] = (_Float16)v;
}

// Pack row-major f32 [K x Nc] into B-fragment tiles:
// tile(kb,nb) base ((kb*(Nc/16)+nb)*512); slot (g*16+n%16)*16 + e, K=16g+e.
__global__ void k_pack_b(const float* __restrict__ s, _Float16* d, int K, int Nc) {
  for (int idx = blockIdx.x * blockDim.x + threadIdx.x; idx < K * Nc;
       idx += gridDim.x * blockDim.x) {
    int k = idx / Nc, n = idx % Nc;
    int kb = k >> 5, kr = k & 31, g = kr >> 4, e = kr & 15;
    int nb = n >> 4, nr = n & 15;
    d[(((size_t)kb * (Nc >> 4) + nb) * 32 + g * 16 + nr) * 16 + e] =
        (_Float16)s[idx];
  }
}

// projB packed per layer: K-dim = d (64), cols = m (288, zero padded), * dn
__global__ void k_projb_pack(const float* __restrict__ proj, _Float16* pb) {
  int idx = blockIdx.x * blockDim.x + threadIdx.x;
  if (idx >= kL * kDim * kMP) return;
  int l = idx / (kDim * kMP);
  int rem = idx - l * kDim * kMP;
  int dd = rem / kMP, m = rem % kMP;
  float v = (m < kM) ? proj[((size_t)l * kM + m) * kDim + dd] * kDn : 0.0f;
  int kb = dd >> 5, kr = dd & 31, g = kr >> 4, e = kr & 15;
  int nb = m >> 4, nr = m & 15;
  pb[(size_t)l * kDim * kMP +
     (((size_t)kb * (kMP >> 4) + nb) * 32 + g * 16 + nr) * 16 + e] = (_Float16)v;
}

// ---------------- GEMM kernels ----------------------------------------------

// C[B*N,512] = h16 @ Wpk + bias ; vmode=1 stores per-head packed-B layout
__global__ void k_gemm_qkv(const _Float16* __restrict__ A,
                           const _Float16* __restrict__ Wpk,
                           const float* __restrict__ bias, _Float16* C, int vmode) {
  int lane = threadIdx.x & 31;
  int wid = blockIdx.x * (blockDim.x >> 5) + (threadIdx.x >> 5);
  int tr = wid >> 4, nbp = wid & 15;                         // 2048 x 16 (32 cols)
  if (tr >= (kB * kN) / 16) return;
  const _Float16* Ap = A + (size_t)tr * 16 * kDim;
  v8f c0 = {}, c1 = {};
#pragma unroll
  for (int kb = 0; kb < 2; ++kb) {
    v16h a = load_a_rm(Ap + kb * 32, kDim, lane);
    const _Float16* bt = Wpk + ((size_t)kb * 32 + nbp * 2) * 512;
    c0 = wmma32(a, load_pk(bt, lane), c0);
    c1 = wmma32(a, load_pk(bt + 512, lane), c1);
  }
  int n = lane & 15, g = lane >> 4;
#pragma unroll
  for (int half = 0; half < 2; ++half) {
    v8f& c = half ? c1 : c0;
    int col = nbp * 32 + half * 16 + n;
    float bb = bias[col];
#pragma unroll
    for (int r = 0; r < 8; ++r) {
      int row = tr * 16 + g * 8 + r;
      float v = c[r] + bb;
      if (!vmode) {
        C[(size_t)row * kInner + col] = (_Float16)v;
      } else {  // v: packed per (b,h) [N x 64] B-fragment layout
        int b = row >> 12, nn = row & (kN - 1);
        int h = col >> 6, d = col & 63;
        int kb2 = nn >> 5, kr = nn & 31, g2 = kr >> 4, e = kr & 15;
        C[(size_t)(b * kH + h) * kN * kDH +
          (((size_t)kb2 * 4 + (d >> 4)) * 32 + g2 * 16 + (d & 15)) * 16 + e] =
            (_Float16)v;
      }
    }
  }
}

// dd[bh][n][m] (row-major f16) = head slice of q/k @ projB_packed
__global__ void k_dd(const _Float16* __restrict__ QK,
                     const _Float16* __restrict__ PB, _Float16* dd) {
  int lane = threadIdx.x & 31;
  int wid = blockIdx.x * (blockDim.x >> 5) + (threadIdx.x >> 5);
  int tr = wid / 9, nbp = wid % 9;                           // 256 x 9 (32 cols)
  if (tr >= kN / 16) return;
  int bh = blockIdx.y, b = bh >> 3, h = bh & 7;
  const _Float16* Ap = QK + ((size_t)b * kN + tr * 16) * kInner + h * kDH;
  v8f c0 = {}, c1 = {};
#pragma unroll
  for (int kb = 0; kb < 2; ++kb) {
    v16h a = load_a_rm(Ap + kb * 32, kInner, lane);
    const _Float16* bt = PB + ((size_t)kb * (kMP >> 4) + nbp * 2) * 512;
    c0 = wmma32(a, load_pk(bt, lane), c0);
    c1 = wmma32(a, load_pk(bt + 512, lane), c1);
  }
  _Float16* Cp = dd + ((size_t)bh * kN + tr * 16) * kMP;
  int n = lane & 15, g = lane >> 4;
#pragma unroll
  for (int half = 0; half < 2; ++half) {
    v8f& c = half ? c1 : c0;
    int col = nbp * 32 + half * 16 + n;
#pragma unroll
    for (int r = 0; r < 8; ++r) Cp[(size_t)(g * 8 + r) * kMP + col] = (_Float16)c[r];
  }
}

__global__ void k_gmax(const _Float16* __restrict__ dd, float* gmax) {
  int bh = blockIdx.x;
  const _Float16* p = dd + (size_t)bh * kN * kMP;
  float mx = -3.0e38f;
  for (size_t i = threadIdx.x; i < (size_t)kN * kMP; i += blockDim.x) {
    int m = (int)(i % kMP);
    if (m < kM) mx = fmaxf(mx, (float)p[i]);
  }
  __shared__ float s[256];
  s[threadIdx.x] = mx;
  __syncthreads();
  for (int o = 128; o > 0; o >>= 1) {
    if ((int)threadIdx.x < o) s[threadIdx.x] = fmaxf(s[threadIdx.x], s[threadIdx.x + o]);
    __syncthreads();
  }
  if (threadIdx.x == 0) gmax[bh] = s[0];
}

// phi = ratio*(exp(dd - diag - mx) + eps)
// kmode=0: row max, write row-major (in-place over dd ok).
// kmode=1: use gmax[bh], write transposed A-fragment packed layout (kp^T).
__global__ void k_phi(const _Float16* __restrict__ dd,
                      const _Float16* __restrict__ D, const float* gmax,
                      _Float16* out, int kmode) {
  int row = blockIdx.x * blockDim.x + threadIdx.x;
  if (row >= kB * kH * kN) return;
  int bh = row / kN, nn = row % kN;
  int b = bh >> 3, h = bh & 7;
  const _Float16* dp = dd + (size_t)row * kMP;
  const _Float16* dr = D + ((size_t)b * kN + nn) * kInner + h * kDH;
  float ss = 0.0f;
#pragma unroll
  for (int d = 0; d < kDH; ++d) {
    float v = (float)dr[d];
    ss += v * v;
  }
  float diag = 0.0625f * ss;                                 // 0.5 * dn^2
  float mx;
  if (kmode) {
    mx = gmax[bh];
  } else {
    mx = -3.0e38f;
    for (int m = 0; m < kM; ++m) mx = fmaxf(mx, (float)dp[m]);
  }
  if (!kmode) {
    _Float16* op = out + (size_t)row * kMP;
    for (int m = 0; m < kMP; ++m) {
      float v = (m < kM) ? kRatio * (__expf((float)dp[m] - diag - mx) + kEps) : 0.0f;
      op[m] = (_Float16)v;
    }
  } else {
    // A^T packing: matrix [kMP x kN], row=m, K=nn. Inverse of A K-map:
    int kb = nn >> 5, kr = nn & 31;
    int g = kr >> 4;
    int e = kr & 15;
    _Float16* base = out + (size_t)bh * kMP * kN;
    for (int m = 0; m < kMP; ++m) {
      float v = (m < kM) ? kRatio * (__expf((float)dp[m] - diag - mx) + kEps) : 0.0f;
      int mb = m >> 4, mr = m & 15;
      base[(((size_t)mb * (kN >> 5) + kb) * 32 + g * 16 + mr) * 16 + e] = (_Float16)v;
    }
  }
}

// kp_sum[bh][m] from packed kp^T
__global__ void k_kpsum(const _Float16* __restrict__ kpt, float* ks) {
  int bh = blockIdx.x, m = threadIdx.x;                      // blockDim = 288
  int mb = m >> 4, mr = m & 15;
  const _Float16* base = kpt + (size_t)bh * kMP * kN;
  float s = 0.0f;
  for (int n = 0; n < kN; ++n) {
    int kb = n >> 5, kr = n & 31;
    int g = (kr >> 3) & 1;
    int e = kr - ((kr < 8) ? 0 : ((kr < 16) ? 8 : ((kr < 24) ? 8 : 16)));
    s += (float)base[(((size_t)mb * (kN >> 5) + kb) * 32 + g * 16 + mr) * 16 + e];
  }
  ks[bh * kMP + m] = s;
}

// ctx = kp^T @ v_head (K=4096): A packed, B packed, wave does 16x64
__global__ void k_ctx(const _Float16* __restrict__ kpt,
                      const _Float16* __restrict__ vpk, _Float16* ctxpk) {
  int lane = threadIdx.x & 31;
  int wid = blockIdx.x * (blockDim.x >> 5) + (threadIdx.x >> 5);
  if (wid >= kMP / 16) return;
  int tr = wid, bh = blockIdx.y;
  const _Float16* Ab = kpt + (size_t)bh * kMP * kN + (size_t)tr * (kN >> 5) * 512;
  const _Float16* Bb = vpk + (size_t)bh * kN * kDH;
  v8f cc[4] = {};
  for (int kb = 0; kb < kN / 32; ++kb) {
    const _Float16* bt = Bb + (size_t)kb * 4 * 512;
    if (kb + 1 < kN / 32) {
      __builtin_prefetch(Ab + (size_t)(kb + 1) * 512, 0, 0);
      __builtin_prefetch(bt + 4 * 512, 0, 0);
    }
    v16h a = load_pk(Ab + (size_t)kb * 512, lane);
#pragma unroll
    for (int tc = 0; tc < 4; ++tc)
      cc[tc] = wmma32(a, load_pk(bt + tc * 512, lane), cc[tc]);
  }
  // store ctx in packed B layout [kMP x 64]
  int n = lane & 15, g = lane >> 4;
  _Float16* base = ctxpk + (size_t)bh * kMP * kDH;
#pragma unroll
  for (int tc = 0; tc < 4; ++tc)
#pragma unroll
    for (int r = 0; r < 8; ++r) {
      int m = tr * 16 + g * 8 + r;
      int kb2 = m >> 5, kr = m & 31, g2 = kr >> 4, e = kr & 15;
      base[(((size_t)kb2 * 4 + tc) * 32 + g2 * 16 + n) * 16 + e] = (_Float16)cc[tc][r];
    }
}

__global__ void k_dinv(const _Float16* __restrict__ qp,
                       const float* __restrict__ ks, float* dinv) {
  int row = blockIdx.x * blockDim.x + threadIdx.x;
  if (row >= kB * kH * kN) return;
  int bh = row / kN;
  const _Float16* qr = qp + (size_t)row * kMP;
  const float* kr = ks + bh * kMP;
  float s = 0.0f;
  for (int m = 0; m < kM; ++m) s += (float)qr[m] * kr[m];
  dinv[row] = 1.0f / s;
}

// attn = d_inv * (qp @ ctx) (K=288): wave does 16x64
__global__ void k_attnout(const _Float16* __restrict__ qp,
                          const _Float16* __restrict__ ctxpk,
                          const float* __restrict__ dinv, _Float16* attn) {
  int lane = threadIdx.x & 31;
  int wid = blockIdx.x * (blockDim.x >> 5) + (threadIdx.x >> 5);
  if (wid >= kN / 16) return;
  int tr = wid, bh = blockIdx.y, b = bh >> 3, h = bh & 7;
  const _Float16* Ap = qp + ((size_t)bh * kN + tr * 16) * kMP;
  const _Float16* Bb = ctxpk + (size_t)bh * kMP * kDH;
  v8f cc[4] = {};
#pragma unroll
  for (int kb = 0; kb < kMP / 32; ++kb) {
    v16h a = load_a_rm(Ap + kb * 32, kMP, lane);
    const _Float16* bt = Bb + (size_t)kb * 4 * 512;
#pragma unroll
    for (int tc = 0; tc < 4; ++tc)
      cc[tc] = wmma32(a, load_pk(bt + tc * 512, lane), cc[tc]);
  }
  int n = lane & 15, g = lane >> 4;
#pragma unroll
  for (int tc = 0; tc < 4; ++tc)
#pragma unroll
    for (int r = 0; r < 8; ++r) {
      int rn = tr * 16 + g * 8 + r;
      float v = cc[tc][r] * dinv[(size_t)bh * kN + rn];
      attn[((size_t)b * kN + rn) * kInner + h * kDH + tc * 16 + n] = (_Float16)v;
    }
}

// h += attn @ Wo + bo (K=512): wave does 16x64; refresh h16
__global__ void k_wo(const _Float16* __restrict__ attn,
                     const _Float16* __restrict__ Wopk,
                     const float* __restrict__ bo, float* h32, _Float16* h16) {
  int lane = threadIdx.x & 31;
  int wid = blockIdx.x * (blockDim.x >> 5) + (threadIdx.x >> 5);
  if (wid >= (kB * kN) / 16) return;
  int tr = wid;
  const _Float16* Ap = attn + (size_t)tr * 16 * kInner;
  v8f cc[4] = {};
#pragma unroll
  for (int kb = 0; kb < kInner / 32; ++kb) {
    v16h a = load_a_rm(Ap + kb * 32, kInner, lane);
    const _Float16* bt = Wopk + (size_t)kb * 4 * 512;
#pragma unroll
    for (int tc = 0; tc < 4; ++tc)
      cc[tc] = wmma32(a, load_pk(bt + tc * 512, lane), cc[tc]);
  }
  int n = lane & 15, g = lane >> 4;
#pragma unroll
  for (int tc = 0; tc < 4; ++tc) {
    int col = tc * 16 + n;
    float bb = bo[col];
#pragma unroll
    for (int r = 0; r < 8; ++r) {
      size_t idx = (size_t)(tr * 16 + g * 8 + r) * kDim + col;
      float nv = h32[idx] + cc[tc][r] + bb;
      h32[idx] = nv;
      h16[idx] = (_Float16)nv;
    }
  }
}

__global__ void k_flat(const float* __restrict__ h32, _Float16* flat16) {
  int idx = blockIdx.x * blockDim.x + threadIdx.x;           // 16*N threads
  if (idx >= 16 * kN) return;
  int b = idx / kN, n = idx % kN;
  float s = 0.0f;
  if (b < kB) {
    const float* p = h32 + ((size_t)b * kN + n) * kDim;
#pragma unroll
    for (int d = 0; d < kDim; ++d) s += p[d];
  }
  flat16[idx] = (_Float16)s;
}

// y1 = relu(flat @ W1 + b1); B tiles staged through LDS via TDM (ISA ch.8)
__global__ void k_mlp1(const _Float16* __restrict__ A,
                       const _Float16* __restrict__ Wpk,
                       const float* __restrict__ bias, _Float16* Y) {
  __shared__ _Float16 smem[8 * 1024];                        // 2KB per wave
  int lane = threadIdx.x & 31, wv = threadIdx.x >> 5;
  int nbp = blockIdx.x * (blockDim.x >> 5) + wv;             // 128 col pairs
  if (nbp >= kN / 32) return;
  const _Float16* stage = smem + wv * 1024;
  unsigned lds_base = (unsigned)(wv * 2048);                 // bytes
  v8f c0 = {}, c1 = {};
  for (int kb = 0; kb < kN / 32; ++kb) {
    // 1-D tensor DMA: copy 1024 contiguous f16 (two packed B tiles) to LDS
    unsigned long long ga =
        (unsigned long long)(uintptr_t)(Wpk + ((size_t)kb * 256 + nbp * 2) * 512);
    u32x4 g0;
    g0[0] = 1u;                                              // count=1
    g0[1] = lds_base;                                        // lds_addr
    g0[2] = (unsigned)ga;                                    // global_addr lo
    g0[3] = ((unsigned)(ga >> 32) & 0x01FFFFFFu) | (2u << 30);  // hi | type=2
    u32x8 g1;
    g1[0] = 1u << 16;                                        // data_size=2B
    g1[1] = 0u;                                              // tdim0 lo16 (1<<24)
    g1[2] = 0x0100u | (1u << 16);                            // tdim0 hi, tdim1=1
    g1[3] = 1024u << 16;                                     // tile_dim0=1024
    g1[4] = 1u;                                              // tile_dim1=1
    g1[5] = 1024u;                                           // dim0 stride
    g1[6] = 0u;
    g1[7] = 0u;
    u32x4 g2 = {0u, 0u, 0u, 0u};
    u32x4 g3 = {0u, 0u, 0u, 0u};
    asm volatile("tensor_load_to_lds %0, %1, %2, %3"
                 :
                 : "s"(g0), "s"(g1), "s"(g2), "s"(g3)
                 : "memory");
    asm volatile("s_wait_tensorcnt 0x0" ::: "memory");
    v16h a = load_a_rm(A + kb * 32, kN, lane);
    c0 = wmma32(a, load_pk(stage, lane), c0);
    c1 = wmma32(a, load_pk(stage + 512, lane), c1);
  }
  int n = lane & 15, g = lane >> 4;
#pragma unroll
  for (int half = 0; half < 2; ++half) {
    v8f& c = half ? c1 : c0;
    int col = nbp * 32 + half * 16 + n;
    float bb = bias[col];
#pragma unroll
    for (int r = 0; r < 8; ++r)
      Y[(size_t)(g * 8 + r) * kN + col] = (_Float16)fmaxf(0.0f, c[r] + bb);
  }
}

// out = y1 @ W2 + b2 (f32, rows 0..7 only)
__global__ void k_mlp2(const _Float16* __restrict__ A,
                       const _Float16* __restrict__ Wpk,
                       const float* __restrict__ bias, float* out) {
  int lane = threadIdx.x & 31;
  int nbp = blockIdx.x * (blockDim.x >> 5) + (threadIdx.x >> 5);
  if (nbp >= kN / 32) return;
  v8f c0 = {}, c1 = {};
  for (int kb = 0; kb < kN / 32; ++kb) {
    const _Float16* bt = Wpk + ((size_t)kb * 256 + nbp * 2) * 512;
    if (kb + 1 < kN / 32) __builtin_prefetch(bt + 256 * 512, 0, 0);
    v16h a = load_a_rm(A + kb * 32, kN, lane);
    c0 = wmma32(a, load_pk(bt, lane), c0);
    c1 = wmma32(a, load_pk(bt + 512, lane), c1);
  }
  int n = lane & 15, g = lane >> 4;
  if (g != 0) return;                                        // rows 8..15 discarded
#pragma unroll
  for (int half = 0; half < 2; ++half) {
    v8f& c = half ? c1 : c0;
    int col = nbp * 32 + half * 16 + n;
    float bb = bias[col];
#pragma unroll
    for (int r = 0; r < 8; ++r) out[(size_t)r * kN + col] = c[r] + bb;
  }
}

// ---------------- launch ----------------------------------------------------

extern "C" void kernel_launch(void* const* d_in, const int* in_sizes, int n_in,
                              void* d_out, int out_size, void* d_ws,
                              size_t ws_size, hipStream_t stream) {
  (void)in_sizes; (void)n_in; (void)out_size; (void)ws_size;
  const float* x    = (const float*)d_in[0];
  const float* Wq   = (const float*)d_in[1];
  const float* bq   = (const float*)d_in[2];
  const float* Wk   = (const float*)d_in[3];
  const float* bk   = (const float*)d_in[4];
  const float* Wv   = (const float*)d_in[5];
  const float* bv   = (const float*)d_in[6];
  const float* Wo   = (const float*)d_in[7];
  const float* bo   = (const float*)d_in[8];
  const float* proj = (const float*)d_in[9];
  const float* W1   = (const float*)d_in[10];
  const float* b1   = (const float*)d_in[11];
  const float* W2   = (const float*)d_in[12];
  const float* b2   = (const float*)d_in[13];
  float* out = (float*)d_out;

  char* wp = (char*)d_ws;
  auto alloc = [&](size_t bytes) -> char* {
    char* r = wp;
    wp += (bytes + 255) & ~(size_t)255;
    return r;
  };
  float*    h32   = (float*)   alloc((size_t)kB * kN * kDim * 4);
  _Float16* h16   = (_Float16*)alloc((size_t)kB * kN * kDim * 2);
  _Float16* qh    = (_Float16*)alloc((size_t)kB * kN * kInner * 2);
  _Float16* kh    = (_Float16*)alloc((size_t)kB * kN * kInner * 2);
  _Float16* vpk   = (_Float16*)alloc((size_t)kB * kH * kN * kDH * 2);
  _Float16* qp    = (_Float16*)alloc((size_t)kB * kH * kN * kMP * 2); // dd_q/qp
  _Float16* ddk   = (_Float16*)alloc((size_t)kB * kH * kN * kMP * 2); // dd_k
  _Float16* kpt   = (_Float16*)alloc((size_t)kB * kH * kN * kMP * 2); // packed kp^T
  float*    gmax  = (float*)   alloc(256);
  float*    ks    = (float*)   alloc((size_t)kB * kH * kMP * 4);
  _Float16* ctxpk = (_Float16*)alloc((size_t)kB * kH * kMP * kDH * 2);
  float*    dinv  = (float*)   alloc((size_t)kB * kH * kN * 4);
  _Float16* attn  = (_Float16*)alloc((size_t)kB * kN * kInner * 2);
  _Float16* flat16= (_Float16*)alloc((size_t)16 * kN * 2);
  _Float16* y1    = (_Float16*)alloc((size_t)16 * kN * 2);
  _Float16* pbpk  = (_Float16*)alloc((size_t)kL * kDim * kMP * 2);
  _Float16* wbig  = (_Float16*)alloc((size_t)kN * kN * 2);
  _Float16* wsm   = (_Float16*)alloc((size_t)kDim * kInner * 2);

  k_init_h<<<(kB * kN * kDim) / 256, 256, 0, stream>>>(x, h32, h16);
  k_projb_pack<<<(kL * kDim * kMP + 255) / 256, 256, 0, stream>>>(proj, pbpk);

  for (int l = 0; l < kL; ++l) {
    const _Float16* pb = pbpk + (size_t)l * kDim * kMP;
    k_pack_b<<<128, 256, 0, stream>>>(Wq + (size_t)l * kDim * kInner, wsm, kDim, kInner);
    k_gemm_qkv<<<4096, 256, 0, stream>>>(h16, wsm, bq + l * kInner, qh, 0);
    k_pack_b<<<128, 256, 0, stream>>>(Wk + (size_t)l * kDim * kInner, wsm, kDim, kInner);
    k_gemm_qkv<<<4096, 256, 0, stream>>>(h16, wsm, bk + l * kInner, kh, 0);
    k_pack_b<<<128, 256, 0, stream>>>(Wv + (size_t)l * kDim * kInner, wsm, kDim, kInner);
    k_gemm_qkv<<<4096, 256, 0, stream>>>(h16, wsm, bv + l * kInner, vpk, 1);

    k_dd<<<dim3(288, kB * kH), 256, 0, stream>>>(qh, pb, qp);
    k_phi<<<(kB * kH * kN) / 256, 256, 0, stream>>>(qp, qh, (const float*)nullptr, qp, 0);
    k_dd<<<dim3(288, kB * kH), 256, 0, stream>>>(kh, pb, ddk);
    k_gmax<<<kB * kH, 256, 0, stream>>>(ddk, gmax);
    k_phi<<<(kB * kH * kN) / 256, 256, 0, stream>>>(ddk, kh, gmax, kpt, 1);

    k_kpsum<<<kB * kH, kMP, 0, stream>>>(kpt, ks);
    k_ctx<<<dim3(3, kB * kH), 256, 0, stream>>>(kpt, vpk, ctxpk);
    k_dinv<<<(kB * kH * kN) / 256, 256, 0, stream>>>(qp, ks, dinv);
    k_attnout<<<dim3(32, kB * kH), 256, 0, stream>>>(qp, ctxpk, dinv, attn);

    k_pack_b<<<128, 256, 0, stream>>>(Wo + (size_t)l * kInner * kDim, wsm, kInner, kDim);
    k_wo<<<256, 256, 0, stream>>>(attn, wsm, bo + l * kDim, h32, h16);
  }

  k_flat<<<(16 * kN) / 256, 256, 0, stream>>>(h32, flat16);
  k_pack_b<<<8192, 256, 0, stream>>>(W1, wbig, kN, kN);
  k_mlp1<<<16, 256, 0, stream>>>(flat16, wbig, b1, y1);
  k_pack_b<<<8192, 256, 0, stream>>>(W2, wbig, kN, kN);
  k_mlp2<<<16, 256, 0, stream>>>(y1, wbig, b2, out);
}
